// CausalSelfAttentionMoH_79353815761465
// MI455X (gfx1250) — matrix operand order repaired and checked by reading, not compiled, not run on hardware
//
#include <hip/hip_runtime.h>
#include <hip/hip_bf16.h>
#include <math.h>

// ---------------------------------------------------------------------------
// CausalSelfAttentionMoH for MI455X (gfx1250, wave32, WMMA bf16 path)
// B=4 T=2048 C=1024 NH=16 NKV=8 HD=64 K_ACTIVE=8 VG=32
// ---------------------------------------------------------------------------

typedef __bf16 bf16;
typedef __bf16 v16bf __attribute__((ext_vector_type(16)));
typedef __bf16 v8bf  __attribute__((ext_vector_type(8)));
typedef float  v8f   __attribute__((ext_vector_type(8)));

#define B_    4
#define T_    2048
#define C_    1024
#define NH_   16
#define NKV_  8
#define HD_   64
#define KACT_ 8
#define VG_   32

// A-matrix fragment (16-bit 16x32, ISA 7.12.2 A table): lane (r, half) holds
// row r, K = 8*half + [0..7] and 16 + 8*half + [0..7]. Caller pre-offsets p by
// 8*half; chunk0 = p[0..7], chunk1 = p[16..23]. Two b128 loads.
__device__ __forceinline__ v16bf load_fragA(const bf16* p) {
  v8bf a = *(const v8bf*)(p);
  v8bf b = *(const v8bf*)(p + 16);
  v16bf r;
#pragma unroll
  for (int i = 0; i < 8; ++i) { r[i] = a[i]; r[i + 8] = b[i]; }
  return r;
}

// B-matrix fragment (16-bit 32x16): lane (r, half) holds column r,
// K = 8*half + [0..7] and 16 + 8*half + [0..7]. Caller pre-offsets p by
// 8*half; chunk0 = p[0..7], chunk1 = p[16..23].
__device__ __forceinline__ v16bf load_fragB(const bf16* p) {
  return load_fragA(p);
}

// ---------------------------------------------------------------------------
// fp32 -> bf16 conversion (weights)
// ---------------------------------------------------------------------------
__global__ void f32_to_bf16_kernel(const float* __restrict__ src,
                                   bf16* __restrict__ dst, int n) {
  int i = blockIdx.x * blockDim.x + threadIdx.x;
  if (i < n) dst[i] = (bf16)src[i];
}

// ---------------------------------------------------------------------------
// Per-token: router logits (fp32), top-8 head mask (exact tie semantics),
// value-gate = 2*sigmoid(x[:32] @ Wg^T), and x -> bf16. One wave per token.
// ---------------------------------------------------------------------------
__global__ void router_kernel(const float* __restrict__ x,
                              const float* __restrict__ Wr,
                              const float* __restrict__ Wg,
                              bf16*  __restrict__ xb,
                              float* __restrict__ hm,
                              float* __restrict__ hm_out,
                              float* __restrict__ gate) {
  const int tok  = blockIdx.x;         // 0 .. B*T-1
  const int lane = threadIdx.x;        // 0 .. 31 (wave32)
  const float* xr = x + (size_t)tok * C_;

  // x row -> bf16
  for (int c = lane; c < C_; c += 32)
    xb[(size_t)tok * C_ + c] = (bf16)xr[c];

  // router logits in fp32 (top-k is precision sensitive)
  float acc[NH_];
#pragma unroll
  for (int h = 0; h < NH_; ++h) acc[h] = 0.f;
  for (int c = lane; c < C_; c += 32) {
    const float xv = xr[c];
#pragma unroll
    for (int h = 0; h < NH_; ++h) acc[h] += xv * Wr[h * C_ + c];
  }
#pragma unroll
  for (int h = 0; h < NH_; ++h) {
#pragma unroll
    for (int m = 16; m >= 1; m >>= 1) acc[h] += __shfl_xor(acc[h], m, 32);
  }

  // threshold = 8th largest value; mask = logits >= threshold (matches jnp)
  {
    const int h = lane & 15;
    float Lh = 0.f;
#pragma unroll
    for (int j = 0; j < NH_; ++j) Lh = (h == j) ? acc[j] : Lh;
    int rank = 0;
#pragma unroll
    for (int j = 0; j < NH_; ++j) rank += (acc[j] > Lh) ? 1 : 0;
    float cand = (rank < KACT_) ? Lh : INFINITY;
#pragma unroll
    for (int m = 8; m >= 1; m >>= 1) cand = fminf(cand, __shfl_xor(cand, m, 32));
    if (lane < NH_) {
      const float mk = (Lh >= cand) ? 1.f : 0.f;
      hm[(size_t)tok * NH_ + h]     = mk;
      hm_out[(size_t)tok * NH_ + h] = mk;
    }
  }

  // value gate
  if (lane < NKV_) {
    float g = 0.f;
#pragma unroll
    for (int c = 0; c < VG_; ++c) g += xr[c] * Wg[lane * VG_ + c];
    gate[(size_t)tok * NKV_ + lane] = 2.f / (1.f + expf(-g));
  }
}

// ---------------------------------------------------------------------------
// C = A(bf16, MxK) @ W(bf16, NxK)^T -> fp32 MxN, K compile-time constant so
// B-row selection folds into 24-bit VMEM immediates (no address spills).
// One wave per 32x64 output tile: 2 A frags x 4 B frags -> 8 WMMAs / k-step.
// ---------------------------------------------------------------------------
template <int K>
__global__ void gemm_bf16_wmma_kernel(const bf16* __restrict__ A,
                                      const bf16* __restrict__ W,
                                      float* __restrict__ Cout,
                                      int M, int N) {
  const int lane = threadIdx.x & 31;
  const int wid  = threadIdx.x >> 5;
  const int r    = lane & 15;
  const int half = lane >> 4;
  const int m32  = M >> 5;
  const int n64  = N >> 6;
  const int w = blockIdx.x * (blockDim.x >> 5) + wid;
  if (w >= m32 * n64) return;
  const int mt = w % m32;
  const int nt = w / m32;

  const bf16* ap0 = A + (size_t)(mt * 32 + r) * K + 8 * half;
  const bf16* ap1 = A + (size_t)(mt * 32 + 16 + r) * K + 8 * half;
  const bf16* wp  = W + (size_t)(nt * 64 + r) * K + 8 * half;

  v8f acc0[4] = {};
  v8f acc1[4] = {};
  for (int k0 = 0; k0 < K; k0 += 32) {
    const v16bf a0 = load_fragA(ap0 + k0);
    const v16bf a1 = load_fragA(ap1 + k0);
    __builtin_prefetch(ap0 + k0 + 128, 0, 1);   // global_prefetch_b8
#pragma unroll
    for (int j = 0; j < 4; ++j) {
      const v16bf b = load_fragB(wp + 16 * j * K + k0);   // imm offset: j*32KB
      acc0[j] = __builtin_amdgcn_wmma_f32_16x16x32_bf16(
          false, a0, false, b, (short)0, acc0[j], false, false);
      acc1[j] = __builtin_amdgcn_wmma_f32_16x16x32_bf16(
          false, a1, false, b, (short)0, acc1[j], false, false);
    }
  }
#pragma unroll
  for (int j = 0; j < 4; ++j)
#pragma unroll
    for (int i = 0; i < 8; ++i) {
      Cout[(size_t)(mt * 32 + i + 8 * half) * N + nt * 64 + 16 * j + r]      = acc0[j][i];
      Cout[(size_t)(mt * 32 + 16 + i + 8 * half) * N + nt * 64 + 16 * j + r] = acc1[j][i];
    }
}

// ---------------------------------------------------------------------------
// Rotary + RMS-norm on a 64-wide head row, fp32 in -> bf16 out, with an
// optional output scale (q is pre-scaled by 1/sqrt(HD)=0.125: exact in bf16,
// removes all score scaling from the attention inner loop).
// One wave per (b,t,h); lane d pairs (d, d+32).
// ---------------------------------------------------------------------------
__global__ void rope_rms_kernel(const float* __restrict__ Qf,
                                const float* __restrict__ cosT,
                                const float* __restrict__ sinT,
                                bf16* __restrict__ Qb, int nh, float oscale) {
  const int w    = blockIdx.x * (blockDim.x >> 5) + (threadIdx.x >> 5);
  const int lane = threadIdx.x & 31;
  if (w >= B_ * T_ * nh) return;
  const int t = (w / nh) % T_;
  const size_t base = (size_t)w * HD_;

  const float x1 = Qf[base + lane];
  const float x2 = Qf[base + 32 + lane];
  const float c  = cosT[(size_t)t * 32 + lane];
  const float s  = sinT[(size_t)t * 32 + lane];
  const float r1 =  x1 * c + x2 * s;
  const float r2 = -x1 * s + x2 * c;

  float ss = r1 * r1 + r2 * r2;
#pragma unroll
  for (int m = 16; m >= 1; m >>= 1) ss += __shfl_xor(ss, m, 32);
  const float sc = rsqrtf(ss * (1.f / HD_) + 1e-6f) * oscale;

  Qb[base + lane]      = (bf16)(r1 * sc);
  Qb[base + 32 + lane] = (bf16)(r2 * sc);
}

// ---------------------------------------------------------------------------
// v = v + gate * ve, stored transposed (d-major) in bf16 so the attention
// kernel's V B-fragments are contiguous loads: vT[(b,kv,d), t].
// ---------------------------------------------------------------------------
__global__ void vgate_kernel(const float* __restrict__ Vf,
                             const float* __restrict__ ve,
                             const float* __restrict__ gate,
                             bf16* __restrict__ vT) {
  const int i = blockIdx.x * blockDim.x + threadIdx.x;
  if (i >= B_ * T_ * NKV_ * HD_) return;
  const int d  = i & (HD_ - 1);
  const int kv = (i >> 6) & (NKV_ - 1);
  const int t  = (i >> 9) & (T_ - 1);
  const int b  = i >> 20;
  const float g = gate[(size_t)(b * T_ + t) * NKV_ + kv];
  const float val = Vf[i] + g * ve[i];
  vT[(((size_t)(b * NKV_ + kv) * HD_) + d) * T_ + t] = (bf16)val;
}

// ---------------------------------------------------------------------------
// Causal GQA flash attention. One wave per (b, h, 16-query tile).
//
// RMS-normed q (pre-scaled by 0.125) and k bound every score to |s| <= ~8, so
// softmax uses a STATIC shift of 8 (shift-invariant, exp arg in [-16.1, 0.1]):
// no running max, no alpha rescaling, no cross-lane reductions in the loop.
// __expf (bare v_exp_f32) suffices: P is rounded to bf16 right after.
// Per-lane partial row sums are butterfly-reduced once at the end.
// Causal split: q0/32 unmasked key blocks + exactly one masked tail block.
// ---------------------------------------------------------------------------
__global__ void flash_attn_kernel(const bf16* __restrict__ Qb,
                                  const bf16* __restrict__ Kb,
                                  const bf16* __restrict__ vT,
                                  const float* __restrict__ hm,
                                  bf16* __restrict__ Yb) {
  __shared__ __align__(128) bf16 pl[8][16][32];   // per-wave 16x32 P tile
  const int lane = threadIdx.x & 31;
  const int wid  = threadIdx.x >> 5;
  const int r    = lane & 15;
  const int half = lane >> 4;
  const int QT   = T_ / 16;                       // 128 query tiles
  const int w = blockIdx.x * (blockDim.x >> 5) + wid;
  if (w >= B_ * NH_ * QT) return;
  const int qt = w % QT;
  const int h  = (w / QT) % NH_;
  const int b  = w / (QT * NH_);
  const int kv = h >> 1;                          // GQA: rep = NH/NKV = 2
  const int q0 = qt * 16;

  const bf16* qrow = Qb + ((size_t)(b * T_ + q0 + r) * NH_ + h) * HD_;
  const v16bf a0 = load_fragA(qrow + 8 * half);
  const v16bf a1 = load_fragA(qrow + 32 + 8 * half);

  const bf16* kbase = Kb + ((size_t)(b * T_ + r) * NKV_ + kv) * HD_ + 8 * half;
  const bf16* vbase = vT + (((size_t)(b * NKV_ + kv) * HD_) + r) * T_ + 8 * half;

  v8f o[4] = {};
  float lsum[8] = {};

  const int nfull = q0 >> 5;                      // fully-unmasked 32-key blocks
  for (int kb = 0; kb < nfull * 32; kb += 32) {
    v8f s0 = {}, s1 = {};
    {
      const bf16* krow = kbase + (size_t)kb * (NKV_ * HD_);
      const v16bf b0 = load_fragB(krow);
      const v16bf b1 = load_fragB(krow + 32);
      s0 = __builtin_amdgcn_wmma_f32_16x16x32_bf16(false, a0, false, b0, (short)0, s0, false, false);
      s0 = __builtin_amdgcn_wmma_f32_16x16x32_bf16(false, a1, false, b1, (short)0, s0, false, false);
    }
    {
      const bf16* krow = kbase + (size_t)(kb + 16) * (NKV_ * HD_);
      const v16bf b0 = load_fragB(krow);
      const v16bf b1 = load_fragB(krow + 32);
      s1 = __builtin_amdgcn_wmma_f32_16x16x32_bf16(false, a0, false, b0, (short)0, s1, false, false);
      s1 = __builtin_amdgcn_wmma_f32_16x16x32_bf16(false, a1, false, b1, (short)0, s1, false, false);
    }
#pragma unroll
    for (int i = 0; i < 8; ++i) {
      const float e0 = __expf(s0[i] - 8.f);
      const float e1 = __expf(s1[i] - 8.f);
      lsum[i] += e0 + e1;
      pl[wid][i + 8 * half][r]      = (bf16)e0;
      pl[wid][i + 8 * half][16 + r] = (bf16)e1;
    }
    const v16bf pa = load_fragA(&pl[wid][r][8 * half]);
#pragma unroll
    for (int j = 0; j < 4; ++j) {
      const v16bf vb = load_fragB(vbase + (size_t)(16 * j) * T_ + kb);
      o[j] = __builtin_amdgcn_wmma_f32_16x16x32_bf16(false, pa, false, vb, (short)0, o[j], false, false);
    }
  }

  // masked tail block (kb = 32*nfull covers keys up to q0+15)
  {
    const int kb = nfull * 32;
    v8f s0 = {}, s1 = {};
    {
      const bf16* krow = kbase + (size_t)kb * (NKV_ * HD_);
      const v16bf b0 = load_fragB(krow);
      const v16bf b1 = load_fragB(krow + 32);
      s0 = __builtin_amdgcn_wmma_f32_16x16x32_bf16(false, a0, false, b0, (short)0, s0, false, false);
      s0 = __builtin_amdgcn_wmma_f32_16x16x32_bf16(false, a1, false, b1, (short)0, s0, false, false);
    }
    {
      const bf16* krow = kbase + (size_t)(kb + 16) * (NKV_ * HD_);
      const v16bf b0 = load_fragB(krow);
      const v16bf b1 = load_fragB(krow + 32);
      s1 = __builtin_amdgcn_wmma_f32_16x16x32_bf16(false, a0, false, b0, (short)0, s1, false, false);
      s1 = __builtin_amdgcn_wmma_f32_16x16x32_bf16(false, a1, false, b1, (short)0, s1, false, false);
    }
#pragma unroll
    for (int i = 0; i < 8; ++i) {
      const int qi = q0 + i + 8 * half;
      const float e0 = (kb + r      > qi) ? 0.f : __expf(s0[i] - 8.f);
      const float e1 = (kb + 16 + r > qi) ? 0.f : __expf(s1[i] - 8.f);
      lsum[i] += e0 + e1;
      pl[wid][i + 8 * half][r]      = (bf16)e0;
      pl[wid][i + 8 * half][16 + r] = (bf16)e1;
    }
    const v16bf pa = load_fragA(&pl[wid][r][8 * half]);
#pragma unroll
    for (int j = 0; j < 4; ++j) {
      const v16bf vb = load_fragB(vbase + (size_t)(16 * j) * T_ + kb);
      o[j] = __builtin_amdgcn_wmma_f32_16x16x32_bf16(false, pa, false, vb, (short)0, o[j], false, false);
    }
  }

  // one butterfly per row at the end (only cross-lane reduction in the kernel)
#pragma unroll
  for (int i = 0; i < 8; ++i) {
#pragma unroll
    for (int m = 8; m >= 1; m >>= 1) lsum[i] += __shfl_xor(lsum[i], m, 32);
  }

  // epilogue: 1/l, head mask, bf16 store for the output GEMM
#pragma unroll
  for (int i = 0; i < 8; ++i) {
    const int q = q0 + i + 8 * half;
    const float msk = hm[(size_t)(b * T_ + q) * NH_ + h];
    const float inv = msk / lsum[i];
#pragma unroll
    for (int j = 0; j < 4; ++j)
      Yb[((size_t)(b * T_ + q) * NH_ + h) * HD_ + 16 * j + r] = (bf16)(o[j][i] * inv);
  }
}

// ---------------------------------------------------------------------------
// Host-side orchestration
// ---------------------------------------------------------------------------
extern "C" void kernel_launch(void* const* d_in, const int* in_sizes, int n_in,
                              void* d_out, int out_size, void* d_ws, size_t ws_size,
                              hipStream_t stream) {
  (void)in_sizes; (void)n_in; (void)out_size; (void)ws_size;
  const float* x    = (const float*)d_in[0];
  const float* ve   = (const float*)d_in[1];
  const float* cosT = (const float*)d_in[2];
  const float* sinT = (const float*)d_in[3];
  const float* Wq   = (const float*)d_in[4];
  const float* Wk   = (const float*)d_in[5];
  const float* Wv   = (const float*)d_in[6];
  const float* Wo   = (const float*)d_in[7];
  const float* Wr   = (const float*)d_in[8];
  const float* Wg   = (const float*)d_in[9];

  float* out      = (float*)d_out;
  float* mask_out = out + (size_t)B_ * T_ * C_;

  const int BT = B_ * T_;
  char* ws = (char*)d_ws;
  size_t off = 0;
  auto alloc = [&](size_t bytes) -> void* {
    void* p = ws + off;
    off = (off + bytes + 255) & ~(size_t)255;
    return p;
  };

  bf16*  xb   = (bf16*) alloc((size_t)BT * C_ * 2);
  bf16*  Wqb  = (bf16*) alloc((size_t)NH_ * HD_ * C_ * 2);
  bf16*  Wkb  = (bf16*) alloc((size_t)NKV_ * HD_ * C_ * 2);
  bf16*  Wvb  = (bf16*) alloc((size_t)NKV_ * HD_ * C_ * 2);
  bf16*  Wob  = (bf16*) alloc((size_t)C_ * C_ * 2);
  float* qf   = (float*)alloc((size_t)BT * NH_ * HD_ * 4);
  float* kf   = (float*)alloc((size_t)BT * NKV_ * HD_ * 4);
  float* vf   = (float*)alloc((size_t)BT * NKV_ * HD_ * 4);
  bf16*  qb   = (bf16*) alloc((size_t)BT * NH_ * HD_ * 2);
  bf16*  kbb  = (bf16*) alloc((size_t)BT * NKV_ * HD_ * 2);
  bf16*  vT   = (bf16*) alloc((size_t)BT * NKV_ * HD_ * 2);
  float* hm   = (float*)alloc((size_t)BT * NH_ * 4);
  float* gate = (float*)alloc((size_t)BT * NKV_ * 4);
  bf16*  yb   = (bf16*)qf;   // reuse: qf is dead after rope_rms(q)

  // weights -> bf16
  { int n = NH_ * HD_ * C_;  f32_to_bf16_kernel<<<(n + 255) / 256, 256, 0, stream>>>(Wq, Wqb, n); }
  { int n = NKV_ * HD_ * C_; f32_to_bf16_kernel<<<(n + 255) / 256, 256, 0, stream>>>(Wk, Wkb, n); }
  { int n = NKV_ * HD_ * C_; f32_to_bf16_kernel<<<(n + 255) / 256, 256, 0, stream>>>(Wv, Wvb, n); }
  { int n = C_ * C_;         f32_to_bf16_kernel<<<(n + 255) / 256, 256, 0, stream>>>(Wo, Wob, n); }

  // router logits / head mask / gate / x->bf16  (one wave per token)
  router_kernel<<<BT, 32, 0, stream>>>(x, Wr, Wg, xb, hm, mask_out, gate);

  auto launch_gemm = [&](const bf16* A, const bf16* W, float* Cp, int M, int N) {
    const int waves  = (M / 32) * (N / 64);
    const int blocks = (waves + 7) / 8;           // 8 waves / 256-thread block
    gemm_bf16_wmma_kernel<C_><<<blocks, 256, 0, stream>>>(A, W, Cp, M, N);
  };

  // projections (K = C_ = 1024 for all GEMMs)
  launch_gemm(xb, Wqb, qf, BT, NH_ * HD_);
  launch_gemm(xb, Wkb, kf, BT, NKV_ * HD_);
  launch_gemm(xb, Wvb, vf, BT, NKV_ * HD_);

  // rotary + rms-norm -> bf16 (q pre-scaled by 1/sqrt(HD), exact power of 2)
  { int wv = BT * NH_;  rope_rms_kernel<<<(wv + 7) / 8, 256, 0, stream>>>(qf, cosT, sinT, qb,  NH_,  0.125f); }
  { int wv = BT * NKV_; rope_rms_kernel<<<(wv + 7) / 8, 256, 0, stream>>>(kf, cosT, sinT, kbb, NKV_, 1.0f); }

  // v gating + transpose to d-major bf16
  { int n = BT * NKV_ * HD_; vgate_kernel<<<(n + 255) / 256, 256, 0, stream>>>(vf, ve, gate, vT); }

  // flash attention (one wave per (b,h,16-query tile))
  { int wv = B_ * NH_ * (T_ / 16);
    flash_attn_kernel<<<(wv + 7) / 8, 256, 0, stream>>>(qb, kbb, vT, hm, yb); }

  // output projection -> fp32 d_out
  launch_gemm(yb, Wob, out, BT, C_);
}
